// PropSTGformer_71055938945749
// MI455X (gfx1250) — matrix operand order repaired and checked by reading, not compiled, not run on hardware
//
#include <hip/hip_runtime.h>
#include <hip/hip_bf16.h>

// ---------------- model constants ----------------
#define B_  2
#define T_  12
#define N_  4096
#define D_  64
#define H_  4
#define HD_ 16
#define R_  (B_*T_*N_)   // 98304 rows of (B,T,N,D)
#define NT_ (T_*D_)      // 768

typedef __bf16 bf16;
typedef bf16         v16bf __attribute__((ext_vector_type(16)));
typedef float        v8f   __attribute__((ext_vector_type(8)));
typedef unsigned int u32x4 __attribute__((ext_vector_type(4)));

union Frag { v16bf v; u32x4 q[2]; };

// =====================================================================
// Generic bf16 WMMA GEMM:  C[b] = A[b] (MxK, row major) * Bt[b]^T
// Bt is stored TRANSPOSED (Ncols x K row-major) so B fragments are
// contiguous.  Block = 256 thr = 8 waves; block tile 128(M) x 64(N);
// each wave computes a 16x64 strip with 4 v_wmma_f32_16x16x32_bf16.
// B panel (64 x 32 bf16) is DOUBLE-BUFFERED in LDS and staged with the
// CDNA5 async path: global_load_async_to_lds_b128 (ASYNCcnt), so the
// fetch of K-step k+32 overlaps the WMMAs of K-step k.
// gx_scatter: epilogue writes x_loc[1] directly into (b,t,n,d) layout.
// =====================================================================
__global__ __launch_bounds__(256)
void k_wmma_gemm(const bf16* __restrict__ A, const bf16* __restrict__ Bt,
                 const float* __restrict__ bias,
                 float* __restrict__ Cf, bf16* __restrict__ Cbf,
                 int M, int Ncols, int K,
                 long sA, long sB, int relu, int gx_scatter)
{
  __shared__ unsigned short sPanel[2][64 * 32];   // [buf][n][k] bf16, 8 KB

  const int lane  = threadIdx.x & 31;
  const int wave  = threadIdx.x >> 5;
  const int bz    = blockIdx.z;
  const int rowBase = blockIdx.x * 128 + wave * 16;
  const int colBase = blockIdx.y * 64;
  const int mlane = lane & 15;
  const int hi    = lane >> 4;                 // half-wave id

  const bf16* Ab = A  + (long)bz * sA;
  const bf16* Bb = Bt + (long)bz * sB;

  v8f acc[4];
#pragma unroll
  for (int j = 0; j < 4; ++j) acc[j] = (v8f)0.0f;

  const long aRow = (long)(rowBase + mlane) * K;
  // cooperative B-panel staging ids: 64 rows x 4 chunks of 8 bf16 (16B/lane)
  const int  cn   = threadIdx.x >> 2;          // 0..63 (panel row = col of C)
  const int  cc   = threadIdx.x & 3;           // chunk
  const long bRow = (long)(colBase + cn) * K + cc * 8;

  // LDS byte offsets of this thread's 16B staging slot in each buffer
  // (generic LDS addresses carry the LDS offset in bits [31:0])
  const unsigned lds0 =
      (unsigned)(unsigned long long)(const void*)&sPanel[0][cn * 32 + cc * 8];
  const unsigned lds1 =
      (unsigned)(unsigned long long)(const void*)&sPanel[1][cn * 32 + cc * 8];

  // async stage of one 16B chunk of the B panel for K-step k into buffer b
  auto stage = [&](int b, int k) {
    unsigned long long ga = (unsigned long long)(const void*)(Bb + bRow + k);
    unsigned           la = b ? lds1 : lds0;
    asm volatile("global_load_async_to_lds_b128 %0, %1, off"
                 :: "v"(la), "v"(ga) : "memory");
  };

  stage(0, 0);
  asm volatile("s_wait_asynccnt 0x0" ::: "memory");
  __syncthreads();

  int buf = 0;
  for (int k = 0; k < K; k += 32) {
    if (k + 32 < K) stage(buf ^ 1, k + 32);    // overlap next fetch with WMMAs

    Frag fa;
    // 16-bit A 16x32 layout: lanes 0-15 K={0..7,16..23}, lanes 16-31 K={8..15,24..31}
    fa.q[0] = *(const u32x4*)(Ab + aRow + k + hi * 8);
    fa.q[1] = *(const u32x4*)(Ab + aRow + k + 16 + hi * 8);
    if (k + 32 < K)
      __builtin_prefetch((const void*)(Ab + aRow + k + 32), 0, 3);

#pragma unroll
    for (int j = 0; j < 4; ++j) {
      Frag fb;
      // 16-bit B 32x16 layout: lane holds column n=lane%16, K = {0..7,16..23} (+8 for hi half)
      const unsigned short* bp = &sPanel[buf][(j * 16 + mlane) * 32 + hi * 8];
      fb.q[0] = *(const u32x4*)(bp);
      fb.q[1] = *(const u32x4*)(bp + 16);
      acc[j] = __builtin_amdgcn_wmma_f32_16x16x32_bf16(
                   false, fa.v, false, fb.v, (short)0, acc[j], false, false);
    }

    asm volatile("s_wait_asynccnt 0x0" ::: "memory");  // next panel arrived
    __syncthreads();                                   // all waves done with cur
    buf ^= 1;
  }

  // epilogue: C 16x16 f32 layout -> lane%16 = col, VGPR r -> row r (+8 for hi half)
#pragma unroll
  for (int j = 0; j < 4; ++j) {
#pragma unroll
    for (int r = 0; r < 8; ++r) {
      int   m   = r + hi * 8;
      int   row = rowBase + m;
      int   col = colBase + j * 16 + mlane;
      float v   = acc[j][r];
      if (bias) v += bias[col];
      if (relu) v = fmaxf(v, 0.0f);
      long idx;
      if (gx_scatter) {                 // col = t*64+d  ->  (b,t,row,d)
        int t = col >> 6, d = col & 63;
        idx = ((((long)bz * T_ + t) * N_ + row) << 6) + d;
      } else {
        idx = (long)bz * (long)M * Ncols + (long)row * Ncols + col;
      }
      if (Cf)  Cf[idx]  = v;
      if (Cbf) Cbf[idx] = (bf16)v;
    }
  }
}

// ---------------- rank-1 source/dest MLPs: a,b vectors ----------------
__global__ void k_ab(const float* __restrict__ X,
                     const float* sw1, const float* sb1, const float* sw2, const float* sb2,
                     const float* dw1, const float* db1, const float* dw2, const float* db2,
                     float* __restrict__ av, float* __restrict__ bv)
{
  int idx = blockIdx.x * blockDim.x + threadIdx.x;
  if (idx >= B_ * N_) return;
  int b = idx / N_, n = idx % N_;
  const float* x = X + (((long)b * T_ + (T_ - 1)) * N_ + n) * D_;
  float xa[64];
#pragma unroll
  for (int d = 0; d < 64; ++d) xa[d] = x[d];
  float accA = sb2[0], accB = db2[0];
  for (int h = 0; h < 64; ++h) {
    float s1 = sb1[h], s2 = db1[h];
    for (int d = 0; d < 64; ++d) {
      s1 += xa[d] * sw1[d * 64 + h];
      s2 += xa[d] * dw1[d * 64 + h];
    }
    accA += fmaxf(s1, 0.f) * sw2[h];
    accB += fmaxf(s2, 0.f) * dw2[h];
  }
  av[idx] = fmaxf(accA, 0.f);
  bv[idx] = fmaxf(accB, 0.f);
}

// ---------------- masked row softmax -> G (bf16) ----------------
__global__ __launch_bounds__(256)
void k_softmax_g(const float* __restrict__ av, const float* __restrict__ bv,
                 const unsigned char* __restrict__ mask, bf16* __restrict__ G)
{
  __shared__ float red[256];
  __shared__ int   ired[256];
  int blk = blockIdx.x;                 // b*N + i
  int b = blk / N_, i = blk % N_;
  int tid = threadIdx.x;
  float ai = av[blk];
  const float* bb = bv + (long)b * N_;
  const unsigned char* mrow = mask + (long)i * N_;   // mask shared across b
  bf16* grow = G + (long)blk * N_;

  float mx = -3.402823e38f; int any = 0;
  for (int j = tid; j < N_; j += 256)
    if (mrow[j]) { any = 1; mx = fmaxf(mx, ai * bb[j]); }
  red[tid] = mx; ired[tid] = any; __syncthreads();
  for (int s = 128; s > 0; s >>= 1) {
    if (tid < s) { red[tid] = fmaxf(red[tid], red[tid + s]); ired[tid] |= ired[tid + s]; }
    __syncthreads();
  }
  mx = red[0]; any = ired[0]; __syncthreads();

  if (!any) {                            // empty row -> identity
    for (int j = tid; j < N_; j += 256) grow[j] = (bf16)((j == i) ? 1.f : 0.f);
    return;
  }
  float sm = 0.f;
  for (int j = tid; j < N_; j += 256)
    if (mrow[j]) sm += __expf(ai * bb[j] - mx);
  red[tid] = sm; __syncthreads();
  for (int s = 128; s > 0; s >>= 1) {
    if (tid < s) red[tid] += red[tid + s];
    __syncthreads();
  }
  float inv = 1.f / fmaxf(red[0], 1e-12f);
  for (int j = tid; j < N_; j += 256)
    grow[j] = (bf16)(mrow[j] ? __expf(ai * bb[j] - mx) * inv : 0.f);
}

// ------- X -> bf16 copy, x_glo = 2X init, XpT (Bt operand of G@X) -------
__global__ void k_convert_X(const float* __restrict__ X, bf16* __restrict__ Xbf,
                            float* __restrict__ xglo, bf16* __restrict__ XpT)
{
  long idx = (long)blockIdx.x * blockDim.x + threadIdx.x;
  if (idx >= (long)R_ * D_) return;
  float v = X[idx];
  Xbf[idx]  = (bf16)v;
  xglo[idx] = 2.f * v;                  // x + x_glo starts at 2X
  long d = idx & 63;
  long n = (idx >> 6) % N_;
  long t = ((idx >> 6) / N_) % T_;
  long b = idx / ((long)T_ * N_ * D_);
  XpT[(b * NT_ + t * 64 + d) * N_ + n] = (bf16)v;   // [b][col][k]
}

// ---------------- weight pack: f32 (K x Nc) -> bf16 (Nc x K) ----------------
__global__ void k_pack_w(const float* __restrict__ src, bf16* __restrict__ dst, int K, int Nc)
{
  int idx = blockIdx.x * blockDim.x + threadIdx.x;
  if (idx >= K * Nc) return;
  int kk = idx / Nc, n = idx % Nc;
  dst[n * K + kk] = (bf16)src[idx];
}

// ---------------- zero fill ----------------
__global__ void k_zero(float* __restrict__ p, long n)
{
  long idx = (long)blockIdx.x * blockDim.x + threadIdx.x;
  if (idx < n) p[idx] = 0.f;
}

// ---------------- per-head l2norm of q,k (in place) ----------------
__global__ void k_l2norm_qk(float* __restrict__ qkv)
{
  int idx = blockIdx.x * blockDim.x + threadIdx.x;
  if (idx >= R_ * H_) return;
  long row = (long)(idx >> 2); int h = idx & 3;
  float* q = qkv + row * 192 + h * 16;
  float* k = qkv + row * 192 + 64 + h * 16;
  float sq = 0.f, sk = 0.f;
#pragma unroll
  for (int m = 0; m < 16; ++m) { sq += q[m] * q[m]; sk += k[m] * k[m]; }
  float iq = 1.f / fmaxf(sqrtf(sq), 1e-12f);
  float ik = 1.f / fmaxf(sqrtf(sk), 1e-12f);
#pragma unroll
  for (int m = 0; m < 16; ++m) { q[m] *= iq; k[m] *= ik; }
}

// ------- spatial linear attention reduction: kvs = k^T v, ks_sum -------
// N split in CH_ chunks for occupancy; partials merged with f32 atomics.
#define CH_ 16
__global__ __launch_bounds__(256)
void k_kvs_partial(const float* __restrict__ qkv, float* __restrict__ kvs,
                   float* __restrict__ kss)
{
  int blk   = blockIdx.x;               // (bt*H + h)*CH_ + chunk
  int chunk = blk % CH_;
  int bh    = blk / CH_;                // bt*H + h
  int bt = bh >> 2, h = bh & 3;
  int m = threadIdx.x >> 4, d = threadIdx.x & 15;
  const float* base = qkv + ((long)bt * N_ + (long)chunk * (N_ / CH_)) * 192;
  float acc = 0.f, ss = 0.f;
  for (int n = 0; n < N_ / CH_; ++n) {
    const float* row = base + (long)n * 192;
    float kf = row[64 + h * 16 + m];
    float vf = row[128 + h * 16 + d];
    acc += kf * vf;
    if (d == 0) ss += kf;
  }
  atomicAdd(&kvs[((long)bh << 8) + m * 16 + d], acc);
  if (d == 0) atomicAdd(&kss[((long)bh << 4) + m], ss);
}

// ---------- spatial apply -> concat cols [0,64) (bf16) ----------
__global__ void k_apply_spatial(const float* __restrict__ qkv, const float* __restrict__ kvs,
                                const float* __restrict__ kss, bf16* __restrict__ concat)
{
  long idx = (long)blockIdx.x * blockDim.x + threadIdx.x;
  if (idx >= (long)R_ * D_) return;
  long row = idx >> 6; int c = (int)(idx & 63);
  int h = c >> 4, d = c & 15;
  long bt = row / N_;
  const float* q  = qkv + row * 192 + h * 16;
  const float* kv = kvs + ((bt * H_ + h) << 8);
  const float* ks = kss + ((bt * H_ + h) << 4);
  float num = 0.f, den = 0.f;
#pragma unroll
  for (int m = 0; m < 16; ++m) { num += q[m] * kv[m * 16 + d]; den += q[m] * ks[m]; }
  float v = qkv[row * 192 + 128 + c];
  num += (float)N_ * v;
  den += (float)N_;
  concat[(row << 7) + c] = (bf16)(num / fmaxf(den, 1e-5f));
}

// ---------- temporal attention (T=12, all in registers) -> concat cols [64,128) ----------
__global__ void k_temporal(const float* __restrict__ qkv, bf16* __restrict__ concat)
{
  long idx = (long)blockIdx.x * blockDim.x + threadIdx.x;
  if (idx >= (long)B_ * N_ * H_ * HD_) return;
  int  d = (int)(idx & 15);
  int  h = (int)((idx >> 4) & 3);
  long n = (idx >> 6) % N_;
  long b = idx / ((long)N_ * 64);
  float kv[16], ks[16];
#pragma unroll
  for (int m = 0; m < 16; ++m) { kv[m] = 0.f; ks[m] = 0.f; }
  for (int t = 0; t < T_; ++t) {
    long row = (b * T_ + t) * N_ + n;
    const float* kr = qkv + row * 192 + 64 + h * 16;
    float vf = qkv[row * 192 + 128 + h * 16 + d];
#pragma unroll
    for (int m = 0; m < 16; ++m) { float kf = kr[m]; kv[m] += kf * vf; ks[m] += kf; }
  }
  for (int t = 0; t < T_; ++t) {
    long row = (b * T_ + t) * N_ + n;
    const float* qr = qkv + row * 192 + h * 16;
    float vf = qkv[row * 192 + 128 + h * 16 + d];
    float num = 0.f, den = 0.f;
#pragma unroll
    for (int m = 0; m < 16; ++m) { num += qr[m] * kv[m]; den += qr[m] * ks[m]; }
    num += (float)T_ * vf;
    den += (float)T_;
    concat[(row << 7) + 64 + h * 16 + d] = (bf16)(num / fmaxf(den, 1e-5f));
  }
}

// ---------------- x_glo += att * pw * scale ----------------
__global__ void k_xglo_acc(float* __restrict__ xglo, const float* __restrict__ att,
                           const float* __restrict__ pw, float scale)
{
  long idx = (long)blockIdx.x * blockDim.x + threadIdx.x;
  if (idx >= (long)R_ * D_) return;
  xglo[idx] += att[idx] * pw[idx] * scale;
}

// ---------------- layernorm over D=64 (optional second addend) ----------------
__global__ void k_layernorm(const float* __restrict__ inA, const float* __restrict__ inB,
                            const float* __restrict__ g, const float* __restrict__ bta,
                            float* __restrict__ outF, bf16* __restrict__ outBf)
{
  long row = (long)blockIdx.x * blockDim.x + threadIdx.x;
  if (row >= R_) return;
  float v[64];
  const float* a  = inA + (row << 6);
  const float* b2 = inB ? inB + (row << 6) : nullptr;
  float mu = 0.f;
#pragma unroll
  for (int d = 0; d < 64; ++d) { float x = a[d] + (b2 ? b2[d] : 0.f); v[d] = x; mu += x; }
  mu *= (1.f / 64.f);
  float var = 0.f;
#pragma unroll
  for (int d = 0; d < 64; ++d) { float e = v[d] - mu; var += e * e; }
  var *= (1.f / 64.f);
  float inv = rsqrtf(var + 1e-5f);
#pragma unroll
  for (int d = 0; d < 64; ++d) {
    float y = (v[d] - mu) * inv * g[d] + bta[d];
    if (outF)  outF[(row << 6) + d]  = y;
    if (outBf) outBf[(row << 6) + d] = (bf16)y;
  }
}

// =====================================================================
extern "C" void kernel_launch(void* const* d_in, const int* in_sizes, int n_in,
                              void* d_out, int out_size, void* d_ws, size_t ws_size,
                              hipStream_t stream)
{
  (void)in_sizes; (void)n_in; (void)out_size; (void)ws_size;

  const float* X      = (const float*)d_in[0];
  const float* src_w1 = (const float*)d_in[1];
  const float* src_b1 = (const float*)d_in[2];
  const float* src_w2 = (const float*)d_in[3];
  const float* src_b2 = (const float*)d_in[4];
  const float* dst_w1 = (const float*)d_in[5];
  const float* dst_b1 = (const float*)d_in[6];
  const float* dst_w2 = (const float*)d_in[7];
  const float* dst_b2 = (const float*)d_in[8];
  const float* qkv_w  = (const float*)d_in[9];    // (2,64,192)
  const float* out_w  = (const float*)d_in[10];   // (2,128,64)
  const float* out_b  = (const float*)d_in[11];   // (2,64)
  const float* pw_w   = (const float*)d_in[12];   // (2,64,64)
  const float* pw_b   = (const float*)d_in[13];   // (2,64)
  const float* ln1_g  = (const float*)d_in[14];
  const float* ln1_b  = (const float*)d_in[15];
  const float* ln2_g  = (const float*)d_in[16];
  const float* ln2_b  = (const float*)d_in[17];
  const float* fc_w1  = (const float*)d_in[18];   // (64,128)
  const float* fc_b1  = (const float*)d_in[19];
  const float* fc_w2  = (const float*)d_in[20];   // (128,64)
  const float* fc_b2  = (const float*)d_in[21];
  const unsigned char* mask = (const unsigned char*)d_in[22];
  float* out = (float*)d_out;

  // ---- workspace carve-up ----
  char* ws = (char*)d_ws;
  size_t off = 0;
  auto alloc = [&](size_t bytes) -> char* {
    char* p = ws + off; off = (off + bytes + 255) & ~(size_t)255; return p;
  };
  float* av     = (float*)alloc((size_t)B_ * N_ * 4);
  float* bvv    = (float*)alloc((size_t)B_ * N_ * 4);
  bf16*  Gbf    = (bf16*)alloc((size_t)B_ * N_ * N_ * 2);
  bf16*  Xbf    = (bf16*)alloc((size_t)R_ * D_ * 2);
  bf16*  XpT    = (bf16*)alloc((size_t)B_ * NT_ * N_ * 2);
  float* xglo   = (float*)alloc((size_t)R_ * D_ * 4);
  bf16*  xloc1  = (bf16*)alloc((size_t)R_ * D_ * 2);
  float* qkvF   = (float*)alloc((size_t)R_ * 192 * 4);
  bf16*  concat = (bf16*)alloc((size_t)R_ * 128 * 2);
  float* kvs    = (float*)alloc((size_t)B_ * T_ * H_ * 256 * 4);
  float* kss    = (float*)alloc((size_t)B_ * T_ * H_ * 16 * 4);
  float* attF   = (float*)alloc((size_t)R_ * D_ * 4);
  bf16*  attB   = (bf16*)alloc((size_t)R_ * D_ * 2);
  float* pwF    = (float*)alloc((size_t)R_ * D_ * 4);
  float* x1F    = (float*)alloc((size_t)R_ * D_ * 4);
  bf16*  x1B    = (bf16*)alloc((size_t)R_ * D_ * 2);
  bf16*  hB     = (bf16*)alloc((size_t)R_ * 128 * 2);
  float* h2F    = (float*)alloc((size_t)R_ * D_ * 4);
  bf16*  qkvT   = (bf16*)alloc((size_t)2 * 192 * 64 * 2);
  bf16*  outT   = (bf16*)alloc((size_t)2 * 64 * 128 * 2);
  bf16*  pwT    = (bf16*)alloc((size_t)2 * 64 * 64 * 2);
  bf16*  fc1T   = (bf16*)alloc((size_t)128 * 64 * 2);
  bf16*  fc2T   = (bf16*)alloc((size_t)64 * 128 * 2);

  auto gemm = [&](const bf16* A, const bf16* Bt, const float* bias,
                  float* Cf, bf16* Cbf, int M, int Nc, int K,
                  long sA, long sB, int batch, int relu, int gx) {
    dim3 grid(M / 128, Nc / 64, batch);
    k_wmma_gemm<<<grid, dim3(256), 0, stream>>>(A, Bt, bias, Cf, Cbf, M, Nc, K,
                                                sA, sB, relu, gx);
  };

  // ---- weight packing (transpose f32 -> bf16) ----
  k_pack_w<<<(64 * 192 + 255) / 256, 256, 0, stream>>>(qkv_w,            qkvT,            64, 192);
  k_pack_w<<<(64 * 192 + 255) / 256, 256, 0, stream>>>(qkv_w + 64 * 192, qkvT + 192 * 64, 64, 192);
  k_pack_w<<<(128 * 64 + 255) / 256, 256, 0, stream>>>(out_w,            outT,            128, 64);
  k_pack_w<<<(128 * 64 + 255) / 256, 256, 0, stream>>>(out_w + 128 * 64, outT + 64 * 128, 128, 64);
  k_pack_w<<<(64 * 64 + 255) / 256, 256, 0, stream>>>(pw_w,              pwT,             64, 64);
  k_pack_w<<<(64 * 64 + 255) / 256, 256, 0, stream>>>(pw_w + 64 * 64,    pwT + 64 * 64,   64, 64);
  k_pack_w<<<(64 * 128 + 255) / 256, 256, 0, stream>>>(fc_w1,            fc1T,            64, 128);
  k_pack_w<<<(128 * 64 + 255) / 256, 256, 0, stream>>>(fc_w2,            fc2T,            128, 64);

  // ---- graph path: a,b -> masked softmax G (bf16) ----
  k_ab<<<(B_ * N_ + 255) / 256, 256, 0, stream>>>(X, src_w1, src_b1, src_w2, src_b2,
                                                  dst_w1, dst_b1, dst_w2, dst_b2, av, bvv);
  k_softmax_g<<<B_ * N_, 256, 0, stream>>>(av, bvv, mask, Gbf);

  // ---- X conversions ----
  long elems = (long)R_ * D_;
  k_convert_X<<<(unsigned)((elems + 255) / 256), 256, 0, stream>>>(X, Xbf, xglo, XpT);

  // ---- x_loc[1] = G @ X   (dominant GEMM, bf16 WMMA, scatter to (b,t,n,d)) ----
  gemm(Gbf, XpT, nullptr, nullptr, xloc1, N_, NT_, N_,
       (long)N_ * N_, (long)NT_ * N_, B_, 0, 1);

  const float SCALE[2] = {1.0f, 0.01f};
  const long  kvsN = (long)B_ * T_ * H_ * 256;
  const long  kssN = (long)B_ * T_ * H_ * 16;
  for (int i = 0; i < 2; ++i) {
    const bf16* zA = (i == 0) ? Xbf : xloc1;
    const bf16* cA = (i == 0) ? Xbf : attB;       // gate input = previous att
    // qkv = z @ w_qkv
    gemm(zA, qkvT + i * 192 * 64, nullptr, qkvF, nullptr, R_, 192, 64, 0, 0, 1, 0, 0);
    k_l2norm_qk<<<(R_ * H_ + 255) / 256, 256, 0, stream>>>(qkvF);
    // linear attention (spatial + temporal) -> concat (bf16, 128 features)
    k_zero<<<(unsigned)((kvsN + kssN + 255) / 256), 256, 0, stream>>>(kvs, kvsN + kssN);
    k_kvs_partial<<<B_ * T_ * H_ * CH_, 256, 0, stream>>>(qkvF, kvs, kss);
    k_apply_spatial<<<(unsigned)((elems + 255) / 256), 256, 0, stream>>>(qkvF, kvs, kss, concat);
    k_temporal<<<(B_ * N_ * H_ * HD_ + 255) / 256, 256, 0, stream>>>(qkvF, concat);
    // pw gate uses PREVIOUS att (cA) -> must run before out-proj overwrites attB
    gemm(cA, pwT + i * 64 * 64, pw_b + i * 64, pwF, nullptr, R_, 64, 64, 0, 0, 1, 0, 0);
    // att = concat @ out_w + out_b
    gemm(concat, outT + i * 64 * 128, out_b + i * 64, attF, attB, R_, 64, 128, 0, 0, 1, 0, 0);
    // x_glo += att * pw * scale
    k_xglo_acc<<<(unsigned)((elems + 255) / 256), 256, 0, stream>>>(xglo, attF, pwF, SCALE[i]);
  }

  // ---- ln1, MLP, ln2 ----
  k_layernorm<<<(R_ + 255) / 256, 256, 0, stream>>>(xglo, nullptr, ln1_g, ln1_b, x1F, x1B);
  gemm(x1B, fc1T, fc_b1, nullptr, hB, R_, 128, 64, 0, 0, 1, 1, 0);   // relu
  gemm(hB, fc2T, fc_b2, h2F, nullptr, R_, 64, 128, 0, 0, 1, 0, 0);
  k_layernorm<<<(R_ + 255) / 256, 256, 0, stream>>>(x1F, h2F, ln2_g, ln2_b, out, nullptr);
}
